// EdgeConvLayer_40269613367298
// MI455X (gfx1250) — matrix operand order repaired and checked by reading, not compiled, not run on hardware
//
#include <hip/hip_runtime.h>

#define DEV __device__ __forceinline__

typedef unsigned short u16;
typedef unsigned int   u32;
typedef __attribute__((ext_vector_type(4)))  u32    u32x4;
typedef __attribute__((ext_vector_type(8)))  float  v8f;
typedef __attribute__((ext_vector_type(16))) __bf16 v16bf;

union Frag { u32x4 q[2]; v16bf v; };

static constexpr int NNODES = 20000;
static constexpr int NEDGES = 320000;
static constexpr int H      = 512;
static constexpr int EDIM   = 64;
static constexpr int KE     = 2 * H + EDIM;   // 1088 (edge concat width)
static constexpr int KN     = 2 * H;          // 1024 (node concat width)

// LDS row strides (u16 units), padded to avoid 64-bank conflicts on b128 reads
static constexpr int FEAT_S = KE + 8;   // 1096 (=137 quads, 548 dwords)
static constexpr int H_S    = H + 8;    // 520
static constexpr int CAT_S  = KN + 8;   // 1032 (=129 quads, 516 dwords)

// ---------- scalar helpers ----------
DEV u16 f2bfbits(float f) {            // fp32 -> bf16 bits, round-to-nearest-even
  u32 u = __float_as_uint(f);
  u32 r = u + 0x7FFFu + ((u >> 16) & 1u);
  return (u16)(r >> 16);
}
DEV u32 packbf(float a, float b) {
  return (u32)f2bfbits(a) | ((u32)f2bfbits(b) << 16);
}
// mish(x) = x*tanh(softplus(x)) = x*((1+e^x)^2-1)/((1+e^x)^2+1)
// one v_exp_f32 + one v_rcp_f32; saturate for large x (tanh->1)
DEV float mishf(float x) {
  float t  = __expf(x);
  float u  = t + 1.0f;
  float u2 = u * u;
  float r  = x * (u2 - 1.0f) * __builtin_amdgcn_rcpf(u2 + 1.0f);
  return (x > 20.0f) ? x : r;
}

// ---------- WMMA fragment helpers (ISA 7.12.2 layouts) ----------
DEV v8f wmma_bf16(v16bf a, v16bf b, v8f c) {
  return __builtin_amdgcn_wmma_f32_16x16x32_bf16(false, a, false, b, (short)0, c,
                                                 false, false);
}
// A 16x32 bf16 tile from LDS (row-major, rowStride in u16 elems)
DEV v16bf load_a(const u16* base, int rowStride, int kBase, int lane) {
  int m  = lane & 15;
  int k0 = (lane < 16) ? 0 : 8;
  const u16* p = base + m * rowStride + kBase + k0;
  Frag f;
  f.q[0] = *(const u32x4*)(p);        // K = kBase+k0   .. +7
  f.q[1] = *(const u32x4*)(p + 16);   // K = kBase+16+k0.. +7
  return f.v;
}
// B 32x16 bf16 tile from pre-blocked weights: blk holds 512 elems laid out
// (n*32 + k), so each lane reads two 8-bf16 runs (k0..k0+7, 16+k0..16+k0+7).
DEV v16bf load_b(const u16* blk, int lane) {
  int n  = lane & 15;
  int k0 = (lane < 16) ? 0 : 8;
  const u16* p = blk + n * 32 + k0;
  Frag f;
  f.q[0] = *(const u32x4*)(p);
  f.q[1] = *(const u32x4*)(p + 16);
  return f.v;
}

// ---------- prep kernels ----------
__global__ __launch_bounds__(256) void cvt_x_kernel(const float* __restrict__ x,
                                                    u16* __restrict__ xb, int n) {
  int i = blockIdx.x * 256 + threadIdx.x;
  if (i < n) xb[i] = f2bfbits(x[i]);
}
// Re-block W[K,512] (row-major fp32) into bf16 blocks:
// dst[((nt*nK + kt)*512) + n*32 + k]  with col = nt*16+n, kRow = kt*32+k
__global__ __launch_bounds__(256) void cvt_w_kernel(const float* __restrict__ W,
                                                    u16* __restrict__ Wb, int K) {
  int i = blockIdx.x * 256 + threadIdx.x;
  if (i >= K * 512) return;
  int kRow = i >> 9, col = i & 511;
  int nt = col >> 4, n = col & 15;
  int kt = kRow >> 5, k = kRow & 31;
  int nK = K >> 5;
  Wb[(size_t)((nt * nK + kt) << 9) + n * 32 + k] = f2bfbits(W[i]);
}
__global__ __launch_bounds__(256) void zero_kernel(float* __restrict__ p, int n) {
  int i = blockIdx.x * 256 + threadIdx.x;
  if (i < n) p[i] = 0.0f;
}

// ---------- fused edge kernel (32 edges/block):
// gather -> GEMM1 -> mish -> GEMM2 -> mish -> atomic scatter ----------
__global__ __launch_bounds__(256) void edge_kernel(
    const u16* __restrict__ xbf, const int* __restrict__ eidx,
    const float* __restrict__ eattr,
    const u16* __restrict__ W1b, const float* __restrict__ b1,
    const u16* __restrict__ W2b, const float* __restrict__ b2,
    float* __restrict__ agg) {
  __shared__ __align__(16) u16 featLds[32 * FEAT_S];  // 70,144 B; reused for h
  __shared__ int sIdx[32], dIdx[32];

  const int tid = threadIdx.x, lane = tid & 31, wave = tid >> 5;
  const int e0 = blockIdx.x * 32;

  if (tid < 32) {
    sIdx[tid] = eidx[e0 + tid];
    dIdx[tid] = eidx[NEDGES + e0 + tid];
  }
  __syncthreads();

  // Stage concat tile [32 x 1088] bf16 into LDS; x parts as 16B quads
  u32x4* featQ = (u32x4*)featLds;              // row stride = 137 quads
  const u32x4* xQ = (const u32x4*)xbf;         // x row = 64 quads
  for (int i = tid; i < 32 * 64; i += 256) {   // x[src] -> cols [0,512)
    int row = i >> 6, c4 = i & 63;
    featQ[row * (FEAT_S / 8) + c4] = xQ[(size_t)sIdx[row] * 64 + c4];
  }
  for (int i = tid; i < 32 * 64; i += 256) {   // x[dst] -> cols [512,1024)
    int row = i >> 6, c4 = i & 63;
    featQ[row * (FEAT_S / 8) + 64 + c4] = xQ[(size_t)dIdx[row] * 64 + c4];
  }
  u32* featU = (u32*)featLds;                  // row stride = 548 dwords
  for (int i = tid; i < 32 * 32; i += 256) {   // edge_attr -> cols [1024,1088)
    int row = i >> 5, c2 = i & 31;
    float a = eattr[(size_t)(e0 + row) * EDIM + c2 * 2];
    float b = eattr[(size_t)(e0 + row) * EDIM + c2 * 2 + 1];
    featU[row * (FEAT_S / 2) + 512 + c2] = packbf(a, b);
  }
  __syncthreads();

  const v8f vzero = {0.f, 0.f, 0.f, 0.f, 0.f, 0.f, 0.f, 0.f};
  const int n = lane & 15, mBase = (lane < 16) ? 0 : 8;

  // GEMM1: [32,1088] x W1 -> h[32,512]; each wave: 2 M-tiles x 4 N-tiles
  v8f acc[2][4];
#pragma unroll
  for (int mt = 0; mt < 2; mt++)
#pragma unroll
    for (int j = 0; j < 4; j++) acc[mt][j] = vzero;
  for (int kt = 0; kt < KE / 32; ++kt) {
    v16bf a0 = load_a(featLds, FEAT_S, kt * 32, lane);
    v16bf a1 = load_a(featLds + 16 * FEAT_S, FEAT_S, kt * 32, lane);
#pragma unroll
    for (int j = 0; j < 4; j++) {
      int nt = wave * 4 + j;
      v16bf bF = load_b(W1b + (size_t)(nt * (KE / 32) + kt) * 512, lane);
      acc[0][j] = wmma_bf16(a0, bF, acc[0][j]);
      acc[1][j] = wmma_bf16(a1, bF, acc[1][j]);
    }
  }
  __syncthreads();                 // feat tile fully consumed -> reuse for h
  u16* hLds = featLds;             // [32 x H_S] = 33,280 B overlay
#pragma unroll
  for (int mt = 0; mt < 2; mt++)
#pragma unroll
    for (int j = 0; j < 4; j++) {
      int col = (wave * 4 + j) * 16 + n;
      float bias = b1[col];
#pragma unroll
      for (int r = 0; r < 8; r++)
        hLds[(mt * 16 + mBase + r) * H_S + col] = f2bfbits(mishf(acc[mt][j][r] + bias));
    }
  __syncthreads();

  // GEMM2: h[32,512] x W2 -> messages, scatter-add into agg[dst]
#pragma unroll
  for (int mt = 0; mt < 2; mt++)
#pragma unroll
    for (int j = 0; j < 4; j++) acc[mt][j] = vzero;
  for (int kt = 0; kt < H / 32; ++kt) {
    v16bf a0 = load_a(hLds, H_S, kt * 32, lane);
    v16bf a1 = load_a(hLds + 16 * H_S, H_S, kt * 32, lane);
#pragma unroll
    for (int j = 0; j < 4; j++) {
      int nt = wave * 4 + j;
      v16bf bF = load_b(W2b + (size_t)(nt * (H / 32) + kt) * 512, lane);
      acc[0][j] = wmma_bf16(a0, bF, acc[0][j]);
      acc[1][j] = wmma_bf16(a1, bF, acc[1][j]);
    }
  }
#pragma unroll
  for (int mt = 0; mt < 2; mt++)
#pragma unroll
    for (int j = 0; j < 4; j++) {
      int col = (wave * 4 + j) * 16 + n;
      float bias = b2[col];
#pragma unroll
      for (int r = 0; r < 8; r++) {
        float msg = mishf(acc[mt][j][r] + bias);
        unsafeAtomicAdd(&agg[(size_t)dIdx[mt * 16 + mBase + r] * H + col], msg);
      }
    }
}

// ---------- fused node kernel (32 nodes/block):
// concat -> GEMM3 -> mish -> GEMM4 -> residual + LayerNorm ----------
__global__ __launch_bounds__(256) void node_kernel(
    const float* __restrict__ xF, const u16* __restrict__ xbf,
    const float* __restrict__ agg,
    const u16* __restrict__ W3b, const float* __restrict__ b3,
    const u16* __restrict__ W4b, const float* __restrict__ b4,
    const float* __restrict__ gamma, const float* __restrict__ beta,
    float* __restrict__ out) {
  __shared__ __align__(16) unsigned char smemRaw[32 * CAT_S * 2]; // 66,048 B
  __shared__ __align__(16) u16 uLds[32 * H_S];                    // 33,280 B
  u16* catLds = (u16*)smemRaw;
  float* outLds = (float*)smemRaw;   // fp32 out tile overlay (65,536 B)

  const int tid = threadIdx.x, lane = tid & 31, wave = tid >> 5;
  const int n0 = blockIdx.x * 32;

  // Stage concat(x_bf16, bf16(agg)) [32 x 1024]
  u32x4* catQ = (u32x4*)catLds;                // row stride = 129 quads
  const u32x4* xQ = (const u32x4*)xbf;
  for (int i = tid; i < 32 * 64; i += 256) {   // x part, coalesced quads
    int row = i >> 6, c4 = i & 63;
    catQ[row * (CAT_S / 8) + c4] = xQ[(size_t)(n0 + row) * 64 + c4];
  }
  u32* catU = (u32*)catLds;                    // row stride = 516 dwords
  for (int i = tid; i < 32 * 256; i += 256) {  // agg part, fp32 -> bf16
    int row = i >> 8, c2 = i & 255;
    float a = agg[(size_t)(n0 + row) * H + c2 * 2];
    float b = agg[(size_t)(n0 + row) * H + c2 * 2 + 1];
    catU[row * (CAT_S / 2) + 256 + c2] = packbf(a, b);
  }
  __syncthreads();

  const v8f vzero = {0.f, 0.f, 0.f, 0.f, 0.f, 0.f, 0.f, 0.f};
  const int n = lane & 15, mBase = (lane < 16) ? 0 : 8;

  // GEMM3: [32,1024] x W3 -> u[32,512], mish
  v8f acc[2][4];
#pragma unroll
  for (int mt = 0; mt < 2; mt++)
#pragma unroll
    for (int j = 0; j < 4; j++) acc[mt][j] = vzero;
  for (int kt = 0; kt < KN / 32; ++kt) {
    v16bf a0 = load_a(catLds, CAT_S, kt * 32, lane);
    v16bf a1 = load_a(catLds + 16 * CAT_S, CAT_S, kt * 32, lane);
#pragma unroll
    for (int j = 0; j < 4; j++) {
      int nt = wave * 4 + j;
      v16bf bF = load_b(W3b + (size_t)(nt * (KN / 32) + kt) * 512, lane);
      acc[0][j] = wmma_bf16(a0, bF, acc[0][j]);
      acc[1][j] = wmma_bf16(a1, bF, acc[1][j]);
    }
  }
#pragma unroll
  for (int mt = 0; mt < 2; mt++)
#pragma unroll
    for (int j = 0; j < 4; j++) {
      int col = (wave * 4 + j) * 16 + n;
      float bias = b3[col];
#pragma unroll
      for (int r = 0; r < 8; r++)
        uLds[(mt * 16 + mBase + r) * H_S + col] = f2bfbits(mishf(acc[mt][j][r] + bias));
    }
  __syncthreads();   // u complete; cat region now dead -> reuse as fp32 out tile

  // GEMM4: u[32,512] x W4 -> out tile (fp32 in LDS for cross-wave LN)
#pragma unroll
  for (int mt = 0; mt < 2; mt++)
#pragma unroll
    for (int j = 0; j < 4; j++) acc[mt][j] = vzero;
  for (int kt = 0; kt < H / 32; ++kt) {
    v16bf a0 = load_a(uLds, H_S, kt * 32, lane);
    v16bf a1 = load_a(uLds + 16 * H_S, H_S, kt * 32, lane);
#pragma unroll
    for (int j = 0; j < 4; j++) {
      int nt = wave * 4 + j;
      v16bf bF = load_b(W4b + (size_t)(nt * (H / 32) + kt) * 512, lane);
      acc[0][j] = wmma_bf16(a0, bF, acc[0][j]);
      acc[1][j] = wmma_bf16(a1, bF, acc[1][j]);
    }
  }
#pragma unroll
  for (int mt = 0; mt < 2; mt++)
#pragma unroll
    for (int j = 0; j < 4; j++) {
      int col = (wave * 4 + j) * 16 + n;
      float bias = b4[col];
#pragma unroll
      for (int r = 0; r < 8; r++)
        outLds[(mt * 16 + mBase + r) * H + col] = acc[mt][j][r] + bias;
    }
  __syncthreads();

  // Residual + LayerNorm: each wave handles 4 rows via shfl_xor reductions
#pragma unroll
  for (int rr = 0; rr < 4; ++rr) {
    int row = wave * 4 + rr;
    size_t node = (size_t)(n0 + row);
    float y[16], s = 0.f, sq = 0.f;
#pragma unroll
    for (int j = 0; j < 16; j++) {
      int col = lane + 32 * j;
      float v = outLds[row * H + col] + xF[node * H + col];
      y[j] = v; s += v; sq += v * v;
    }
#pragma unroll
    for (int off = 16; off; off >>= 1) {
      s += __shfl_xor(s, off, 32);
      sq += __shfl_xor(sq, off, 32);
    }
    float mu = s * (1.0f / H);
    float var = sq * (1.0f / H) - mu * mu;
    float rs = rsqrtf(var + 1e-5f);
#pragma unroll
    for (int j = 0; j < 16; j++) {
      int col = lane + 32 * j;
      out[node * H + col] = (y[j] - mu) * rs * gamma[col] + beta[col];
    }
  }
}

// ---------- host launcher ----------
extern "C" void kernel_launch(void* const* d_in, const int* in_sizes, int n_in,
                              void* d_out, int out_size, void* d_ws, size_t ws_size,
                              hipStream_t stream) {
  (void)in_sizes; (void)n_in; (void)out_size; (void)ws_size;
  const float* x     = (const float*)d_in[0];
  const int*   eidx  = (const int*)d_in[1];     // JAX default x64-off -> int32
  const float* eattr = (const float*)d_in[2];
  const float* W1 = (const float*)d_in[3];  const float* b1 = (const float*)d_in[4];
  const float* W2 = (const float*)d_in[5];  const float* b2 = (const float*)d_in[6];
  const float* W3 = (const float*)d_in[7];  const float* b3 = (const float*)d_in[8];
  const float* W4 = (const float*)d_in[9];  const float* b4 = (const float*)d_in[10];
  const float* gamma = (const float*)d_in[11];
  const float* beta  = (const float*)d_in[12];
  float* out = (float*)d_out;

  // Workspace layout (bytes, all 256-aligned)
  char* ws = (char*)d_ws;
  u16*   xbf = (u16*)(ws + 0);                   // 20,480,000
  u16*   W1b = (u16*)(ws + 20480000);            //  1,114,112
  u16*   W2b = (u16*)(ws + 21594112);            //    524,288
  u16*   W3b = (u16*)(ws + 22118400);            //  1,048,576
  u16*   W4b = (u16*)(ws + 23166976);            //    524,288
  float* agg = (float*)(ws + 23691264);          // 40,960,000  (total ~64.7 MB)

  const int nx = NNODES * H;
  cvt_x_kernel<<<(nx + 255) / 256, 256, 0, stream>>>(x, xbf, nx);
  cvt_w_kernel<<<(KE * 512 + 255) / 256, 256, 0, stream>>>(W1, W1b, KE);
  cvt_w_kernel<<<(H * 512 + 255) / 256, 256, 0, stream>>>(W2, W2b, H);
  cvt_w_kernel<<<(KN * 512 + 255) / 256, 256, 0, stream>>>(W3, W3b, KN);
  cvt_w_kernel<<<(H * 512 + 255) / 256, 256, 0, stream>>>(W4, W4b, H);
  zero_kernel<<<(nx + 255) / 256, 256, 0, stream>>>(agg, nx);

  edge_kernel<<<NEDGES / 32, 256, 0, stream>>>(xbf, eidx, eattr, W1b, b1, W2b, b2, agg);
  node_kernel<<<NNODES / 32, 256, 0, stream>>>(x, xbf, agg, W3b, b3, W4b, b4,
                                               gamma, beta, out);
}
